// MultimodalEncoder_91164975825122
// MI455X (gfx1250) — matrix operand order repaired and checked by reading, not compiled, not run on hardware
//
#include <hip/hip_runtime.h>

typedef __attribute__((ext_vector_type(16))) _Float16 v16h;
typedef __attribute__((ext_vector_type(8)))  float    v8f;

#define NFFN  6
#define GD    32
#define ED    64
#define EPS   1e-5f

// ---------------- argument structs (passed by value) ----------------
struct PrepArgs {
  const float* W1[NFFN];
  const float* W2[NFFN];
  const float* Wf;
  const float* ticker_emb; const float* ticker_g; const float* ticker_b;
  const int*   ticker_id;
  const float* regime_emb; const float* regime_g; const float* regime_b;
  _Float16* wsW1; _Float16* wsW2; _Float16* wsWf; _Float16* wsTk; _Float16* wsRg;
};

struct EncArgs {
  const float* x;
  const float* b1[NFFN]; const float* g1[NFFN]; const float* be1[NFFN];
  const float* b2[NFFN]; const float* g2[NFFN]; const float* be2[NFFN];
  const float* bf; const float* gf; const float* bef;
  const _Float16* wsW1; const _Float16* wsW2; const _Float16* wsWf;
  const _Float16* wsTk; const _Float16* wsRg;
  float* out;
};

// ---------------- helpers ----------------
// 16-bit A-matrix 16x32 K placement (ISA 7.12.2): VGPR r holds K pair (kb, kb+1)
__device__ __host__ __forceinline__ int a_kbase(int r, int hh) {
  return (r < 4) ? (hh * 8 + 2 * r) : (16 + hh * 8 + 2 * (r - 4));
}

__device__ __forceinline__ void lds_fence() {
  __builtin_amdgcn_wave_barrier();
  asm volatile("s_wait_dscnt 0x0" ::: "memory");
  __builtin_amdgcn_wave_barrier();
}

// async global->LDS copy of one dword (gfx1250 path, tracked by ASYNCcnt)
__device__ __forceinline__ void async_ld_b32(unsigned lds_off, const float* gptr) {
  unsigned long long ga = (unsigned long long)(size_t)gptr;
  asm volatile("global_load_async_to_lds_b32 %0, %1, off"
               :: "v"(lds_off), "v"(ga) : "memory");
}
__device__ __forceinline__ void wait_async0() {
  asm volatile("s_wait_asynccnt 0x0" ::: "memory");
  __builtin_amdgcn_wave_barrier();
}

__device__ __forceinline__ v8f ld8(const float* p) { return *(const v8f*)p; }

__device__ __forceinline__ unsigned pkh2(float a, float b) {
  unsigned lo = (unsigned)__builtin_bit_cast(unsigned short, (_Float16)a);
  unsigned hi = (unsigned)__builtin_bit_cast(unsigned short, (_Float16)b);
  return lo | (hi << 16);
}

// ---------------- prep kernel: pack W^T into WMMA *A*-fragment layout -------
// A frag (16x32 f16): lane l holds row M=l&15; element pair (2r,2r+1) = K pair
__global__ void mm_prep_91164975825122(PrepArgs p) {
  const int indim[NFFN] = {6, 5, 8, 4, 2, 1};
  int gid = blockIdx.x * blockDim.x + threadIdx.x;
  int nth = gridDim.x * blockDim.x;

  for (int i = gid; i < NFFN * 2 * 512; i += nth) {   // W1^T, K zero-padded to 32
    int f = i >> 10, rem = i & 1023, nt = rem >> 9, idx = rem & 511;
    int lane = idx >> 4, e = idx & 15, r = e >> 1, s = e & 1;
    int K = a_kbase(r, lane >> 4) + s;
    int M = nt * 16 + (lane & 15);                    // output feature
    float v = (K < indim[f]) ? p.W1[f][K * GD + M] : 0.0f;
    p.wsW1[i] = (_Float16)v;
  }
  for (int i = gid; i < NFFN * 2 * 512; i += nth) {   // W2^T
    int f = i >> 10, rem = i & 1023, nt = rem >> 9, idx = rem & 511;
    int lane = idx >> 4, e = idx & 15, r = e >> 1, s = e & 1;
    int K = (lane >> 4) * 16 + e;
    int M = nt * 16 + (lane & 15);
    p.wsW2[i] = (_Float16)p.W2[f][K * GD + M];
  }
  for (int i = gid; i < 8 * 4 * 512; i += nth) {      // Wf^T: [ktile8][ntile4][frag512]
    int kt = i >> 11, rem = i & 2047, nt = rem >> 9, idx = rem & 511;
    int lane = idx >> 4, e = idx & 15, r = e >> 1, s = e & 1;
    int K = kt * 32 + (lane >> 4) * 16 + e;           // fused feature
    int M = nt * 16 + (lane & 15);                    // output feature
    p.wsWf[i] = (_Float16)p.Wf[K * ED + M];
  }
  for (int b = gid; b < 128; b += nth) {              // ticker rows: LN'd, f16
    const float* row = p.ticker_emb + p.ticker_id[b] * GD;
    float s = 0.f, ss = 0.f;
    for (int j = 0; j < GD; j++) { float e = row[j]; s += e; ss += e * e; }
    float mean = s * (1.0f / GD), var = ss * (1.0f / GD) - mean * mean;
    float inv = rsqrtf(var + EPS);
    for (int j = 0; j < GD; j++)
      p.wsTk[b * GD + j] = (_Float16)((row[j] - mean) * inv * p.ticker_g[j] + p.ticker_b[j]);
  }
  for (int r3 = gid; r3 < 3; r3 += nth) {             // 3 regime rows: LN'd, f16
    const float* row = p.regime_emb + r3 * GD;
    float s = 0.f, ss = 0.f;
    for (int j = 0; j < GD; j++) { float e = row[j]; s += e; ss += e * e; }
    float mean = s * (1.0f / GD), var = ss * (1.0f / GD) - mean * mean;
    float inv = rsqrtf(var + EPS);
    for (int j = 0; j < GD; j++)
      p.wsRg[r3 * GD + j] = (_Float16)((row[j] - mean) * inv * p.regime_g[j] + p.regime_b[j]);
  }
}

// ---------------- main fused encoder (transposed GEMMs) ----------------
// D = W^T(A, M=features) x act^T(B, N=tokens); each token's features live in a
// lane pair -> LayerNorm = intra-lane vector sums + one shfl_xor(16).
__global__ __launch_bounds__(128) void mm_encoder_91164975825122(EncArgs p, int ntiles) {
  __shared__ float    sxs[4][16][28];   // 7 KB : staged x rows (async-filled)
  __shared__ _Float16 shs[4][16][GD];   // 4 KB : token-major activation rows
  const int lane = threadIdx.x & 31;
  const int wv   = threadIdx.x >> 5;
  const int hh   = lane >> 4;           // which feature half this lane owns
  const int lN   = lane & 15;           // token index within tile
  float*    sx = &sxs[wv][0][0];
  _Float16* sh = &shs[wv][0][0];
  unsigned* sh32 = (unsigned*)sh;
  const unsigned sx_base = (unsigned)(size_t)(void*)sx;

  const int lo_[NFFN] = {0, 6, 11, 19, 23, 26};
  const int nd_[NFFN] = {6, 5, 8, 4, 2, 1};
  const int kt_[NFFN] = {0, 1, 2, 3, 4, 6};   // final-GEMM K-tile of each FFN output

  int waveId = blockIdx.x * 4 + wv;
  int nwaves = gridDim.x * 4;

  for (int t = waveId; t < ntiles; t += nwaves) {
    int tok0 = t * 16;
    if (t + nwaves < ntiles)
      __builtin_prefetch(p.x + (size_t)(t + nwaves) * 16 * 27, 0, 1);

    // ---- stage x: 16 tokens x 27 f32, async global->LDS ----
    for (int i = lane; i < 16 * 27; i += 32) {
      int m = i / 27, c = i % 27;
      async_ld_b32(sx_base + (unsigned)(m * 28 + c) * 4u,
                   p.x + (size_t)(tok0 + m) * 27 + c);
    }
    wait_async0();

    v8f acc[4] = {{}, {}, {}, {}};   // final: M=out-feature tiles, N=tokens

    // ---- six FFNs, each immediately folded into the final projection ----
    for (int f = 0; f < NFFN; ++f) {
      int lo = lo_[f], nd = nd_[f];
      // B fragment: x^T, K=input feature (lane's half), N=token=lN
      v16h b = {};
      if (hh == 0) {
#pragma unroll
        for (int e = 0; e < 8; e++)
          b[e] = (e < nd) ? (_Float16)sx[lN * 28 + lo + e] : (_Float16)0.f;
      }
      const _Float16* wf1 = p.wsW1 + f * 1024;
      v16h wa0 = *(const v16h*)(wf1 +   0 + lane * 16);
      v16h wa1 = *(const v16h*)(wf1 + 512 + lane * 16);
      v8f c0 = {}, c1 = {};
      c0 = __builtin_amdgcn_wmma_f32_16x16x32_f16(false, wa0, false, b, (short)0, c0, false, false);
      c1 = __builtin_amdgcn_wmma_f32_16x16x32_f16(false, wa1, false, b, (short)0, c1, false, false);

      // per-lane features: tile0 -> hh*8+0..7, tile1 -> 16+hh*8+0..7
      v8f x0 = c0 + ld8(p.b1[f] + hh * 8);
      v8f x1 = c1 + ld8(p.b1[f] + 16 + hh * 8);
      float s = 0.f, ss = 0.f;
#pragma unroll
      for (int i = 0; i < 8; i++) {
        s  += x0[i] + x1[i];
        ss += x0[i] * x0[i] + x1[i] * x1[i];
      }
      s  += __shfl_xor(s, 16, 32);     // combine the two feature halves
      ss += __shfl_xor(ss, 16, 32);
      float mean = s * (1.0f / GD);
      float var  = ss * (1.0f / GD) - mean * mean;
      float inv  = rsqrtf(var + EPS);
      v8f h0 = (x0 - mean) * inv * ld8(p.g1[f] + hh * 8)      + ld8(p.be1[f] + hh * 8);
      v8f h1 = (x1 - mean) * inv * ld8(p.g1[f] + 16 + hh * 8) + ld8(p.be1[f] + 16 + hh * 8);
      h0 = __builtin_elementwise_max(h0, (v8f)0.f);
      h1 = __builtin_elementwise_max(h1, (v8f)0.f);
      // store token-major: sh[token][feature]; dwords = packed f16 pairs
#pragma unroll
      for (int q = 0; q < 4; q++) {
        sh32[lN * 16 + hh * 4 + q]     = pkh2(h0[2 * q], h0[2 * q + 1]);  // feats hh*8+..
        sh32[lN * 16 + 8 + hh * 4 + q] = pkh2(h1[2 * q], h1[2 * q + 1]);  // feats 16+hh*8+..
      }
      lds_fence();

      // FFN layer 2: B = token-major h row (contiguous 32B)
      v16h b2 = *(const v16h*)(sh + lN * GD + hh * 16);
      const _Float16* wf2 = p.wsW2 + f * 1024;
      v16h wc0 = *(const v16h*)(wf2 +   0 + lane * 16);
      v16h wc1 = *(const v16h*)(wf2 + 512 + lane * 16);
      v8f d0 = {}, d1 = {};
      d0 = __builtin_amdgcn_wmma_f32_16x16x32_f16(false, wc0, false, b2, (short)0, d0, false, false);
      d1 = __builtin_amdgcn_wmma_f32_16x16x32_f16(false, wc1, false, b2, (short)0, d1, false, false);

      v8f y0 = d0 + ld8(p.b2[f] + hh * 8);
      v8f y1 = d1 + ld8(p.b2[f] + 16 + hh * 8);
      s = 0.f; ss = 0.f;
#pragma unroll
      for (int i = 0; i < 8; i++) {
        s  += y0[i] + y1[i];
        ss += y0[i] * y0[i] + y1[i] * y1[i];
      }
      s  += __shfl_xor(s, 16, 32);
      ss += __shfl_xor(ss, 16, 32);
      mean = s * (1.0f / GD);
      var  = ss * (1.0f / GD) - mean * mean;
      inv  = rsqrtf(var + EPS);
      v8f o0 = (y0 - mean) * inv * ld8(p.g2[f] + hh * 8)      + ld8(p.be2[f] + hh * 8);
      v8f o1 = (y1 - mean) * inv * ld8(p.g2[f] + 16 + hh * 8) + ld8(p.be2[f] + 16 + hh * 8);
#pragma unroll
      for (int q = 0; q < 4; q++) {
        sh32[lN * 16 + hh * 4 + q]     = pkh2(o0[2 * q], o0[2 * q + 1]);
        sh32[lN * 16 + 8 + hh * 4 + q] = pkh2(o1[2 * q], o1[2 * q + 1]);
      }
      lds_fence();

      // fold into final projection: this source is K-tile kt_[f]
      v16h bf16 = *(const v16h*)(sh + lN * GD + hh * 16);
      const _Float16* wff = p.wsWf + kt_[f] * 2048;
#pragma unroll
      for (int nt = 0; nt < 4; nt++) {
        v16h w = *(const v16h*)(wff + nt * 512 + lane * 16);
        acc[nt] = __builtin_amdgcn_wmma_f32_16x16x32_f16(false, w, false, bf16, (short)0, acc[nt], false, false);
      }
      lds_fence();
    }

    // ---- regime (K-tile 5): per-token row pick, contiguous 32B B-fragment ----
    {
      int ri = (int)roundf(sx[lN * 28 + 25]);
      ri = ri < 0 ? 0 : (ri > 2 ? 2 : ri);
      v16h b = *(const v16h*)(p.wsRg + ri * GD + hh * 16);
      const _Float16* wff = p.wsWf + 5 * 2048;
#pragma unroll
      for (int nt = 0; nt < 4; nt++) {
        v16h w = *(const v16h*)(wff + nt * 512 + lane * 16);
        acc[nt] = __builtin_amdgcn_wmma_f32_16x16x32_f16(false, w, false, b, (short)0, acc[nt], false, false);
      }
    }
    // ---- ticker (K-tile 7): whole tile shares one batch row ----
    {
      int bix = tok0 >> 12; // tok / 4096
      v16h b = *(const v16h*)(p.wsTk + bix * GD + hh * 16);
      const _Float16* wff = p.wsWf + 7 * 2048;
#pragma unroll
      for (int nt = 0; nt < 4; nt++) {
        v16h w = *(const v16h*)(wff + nt * 512 + lane * 16);
        acc[nt] = __builtin_amdgcn_wmma_f32_16x16x32_f16(false, w, false, b, (short)0, acc[nt], false, false);
      }
    }

    // ---- final bias + LN(64) + wide non-temporal stores ----
    // lane's features: t*16 + hh*8 + 0..7 ; token = lN
    v8f va[4];
#pragma unroll
    for (int tt = 0; tt < 4; tt++) va[tt] = acc[tt] + ld8(p.bf + tt * 16 + hh * 8);
    float s = 0.f, ss = 0.f;
#pragma unroll
    for (int tt = 0; tt < 4; tt++)
#pragma unroll
      for (int i = 0; i < 8; i++) { s += va[tt][i]; ss += va[tt][i] * va[tt][i]; }
    s  += __shfl_xor(s, 16, 32);
    ss += __shfl_xor(ss, 16, 32);
    float mean = s * (1.0f / ED);
    float var  = ss * (1.0f / ED) - mean * mean;
    float inv  = rsqrtf(var + EPS);
    float* outrow = p.out + (size_t)(tok0 + lN) * ED + hh * 8;
#pragma unroll
    for (int tt = 0; tt < 4; tt++) {
      v8f o = (va[tt] - mean) * inv * ld8(p.gf + tt * 16 + hh * 8)
            + ld8(p.bef + tt * 16 + hh * 8);
      __builtin_nontemporal_store(o, (v8f*)(outrow + tt * 16));
    }
    lds_fence();
  }
}

// ---------------- host launch ----------------
extern "C" void kernel_launch(void* const* d_in, const int* in_sizes, int n_in,
                              void* d_out, int out_size, void* d_ws, size_t ws_size,
                              hipStream_t stream) {
  (void)n_in; (void)out_size; (void)ws_size;
  auto F = [&](int i) { return (const float*)d_in[i]; };

  _Float16* ws   = (_Float16*)d_ws;
  _Float16* wsW1 = ws;             // 6144 halves
  _Float16* wsW2 = ws + 6144;      // 6144 halves
  _Float16* wsWf = ws + 12288;     // 16384 halves
  _Float16* wsTk = ws + 28672;     // 4096 halves
  _Float16* wsRg = ws + 32768;     // 96 halves

  PrepArgs pp;
  for (int f = 0; f < NFFN; ++f) {
    pp.W1[f] = F(2 + 8 * f + 0);
    pp.W2[f] = F(2 + 8 * f + 4);
  }
  pp.Wf = F(56);
  pp.ticker_emb = F(53); pp.ticker_g = F(54); pp.ticker_b = F(55);
  pp.ticker_id = (const int*)d_in[1];
  pp.regime_emb = F(50); pp.regime_g = F(51); pp.regime_b = F(52);
  pp.wsW1 = wsW1; pp.wsW2 = wsW2; pp.wsWf = wsWf; pp.wsTk = wsTk; pp.wsRg = wsRg;
  mm_prep_91164975825122<<<32, 256, 0, stream>>>(pp);

  EncArgs ep;
  ep.x = F(0);
  for (int f = 0; f < NFFN; ++f) {
    ep.b1[f]  = F(2 + 8 * f + 1);
    ep.g1[f]  = F(2 + 8 * f + 2);
    ep.be1[f] = F(2 + 8 * f + 3);
    ep.b2[f]  = F(2 + 8 * f + 5);
    ep.g2[f]  = F(2 + 8 * f + 6);
    ep.be2[f] = F(2 + 8 * f + 7);
  }
  ep.bf = F(57); ep.gf = F(58); ep.bef = F(59);
  ep.wsW1 = wsW1; ep.wsW2 = wsW2; ep.wsWf = wsWf; ep.wsTk = wsTk; ep.wsRg = wsRg;
  ep.out = (float*)d_out;

  int ntokens = in_sizes[0] / 27;  // 128*4096
  int ntiles  = ntokens / 16;      // 32768
  mm_encoder_91164975825122<<<2048, 128, 0, stream>>>(ep, ntiles);
}
